// Attention_88545045774775
// MI455X (gfx1250) — matrix operand order repaired and checked by reading, not compiled, not run on hardware
//
#include <hip/hip_runtime.h>
#include <hip/hip_bf16.h>

typedef __attribute__((ext_vector_type(16))) __bf16 v16bf;
typedef __attribute__((ext_vector_type(8)))  __bf16 v8bf;
typedef __attribute__((ext_vector_type(8)))  float  v8f;

#define DIM    1024
#define NHEADS 8
#define DHEAD  64
#define INNER  512
#define SEQ    2048
#define BATCH  2
#define NROWS  (BATCH*SEQ)      /* 4096 */
#define QK_SCALE 0.125f         /* 64^-0.5 */
#define NEG_BIG (-3.0e38f)

// ---------------- fragment loaders (CDNA5 WMMA 16x16x32 bf16 layouts) ----------

// A (16x32, 16-bit): lane%16 = row M, lane/16 selects K-half.
// VGPR0..3 hold K = k0 + 8*half + 0..7 ; VGPR4..7 hold K = k0 + 16 + 8*half + 0..7
__device__ __forceinline__ v16bf load_frag_a(const __bf16* rowbase, int k0, int lane) {
  const __bf16* p = rowbase + k0 + ((lane >> 4) << 3);
  v8bf lo = *(const v8bf*)p;
  v8bf hi = *(const v8bf*)(p + 16);
  v16bf a;
#pragma unroll
  for (int i = 0; i < 8; ++i) { a[i] = lo[i]; a[i + 8] = hi[i]; }
  return a;
}

// B (32x16, 16-bit): lane%16 = col N; VGPR0..3 hold K = k0 + 8*half + 0..7 ; VGPR4..7 hold K = k0 + 16 + 8*half + 0..7
// colbase must point at BT[col*ldk] (K-contiguous storage).
__device__ __forceinline__ v16bf load_frag_bt(const __bf16* colbase, int k0, int lane) {
  const __bf16* p = colbase + k0 + ((lane >> 4) << 3);
  v8bf lo = *(const v8bf*)p;
  v8bf hi = *(const v8bf*)(p + 16);
  v16bf b;
#pragma unroll
  for (int i = 0; i < 8; ++i) { b[i] = lo[i]; b[i + 8] = hi[i]; }
  return b;
}

__device__ __forceinline__ v8f wmma_bf16(v16bf a, v16bf b, v8f c) {
  return __builtin_amdgcn_wmma_f32_16x16x32_bf16(false, a, false, b, (short)0, c, false, false);
}

// reductions across the 16-lane row group (wave32: two row groups per wave)
__device__ __forceinline__ float rowmax16(float x) {
  x = fmaxf(x, __shfl_xor(x, 8, 16));
  x = fmaxf(x, __shfl_xor(x, 4, 16));
  x = fmaxf(x, __shfl_xor(x, 2, 16));
  x = fmaxf(x, __shfl_xor(x, 1, 16));
  return x;
}
__device__ __forceinline__ float rowsum16(float x) {
  x += __shfl_xor(x, 8, 16);
  x += __shfl_xor(x, 4, 16);
  x += __shfl_xor(x, 2, 16);
  x += __shfl_xor(x, 1, 16);
  return x;
}

// ---------------- async global->LDS copy (CDNA5, ASYNCcnt tracked) ------------
// GLOBAL_LOAD_ASYNC_TO_LDS_B128: VDST = VGPR with LDS byte address, VADDR = 64-bit
// global address (GV mode, SADDR=off). LDS aperture rule: lds offset = addr[31:0].

__device__ __forceinline__ void async_copy16(const __bf16* g, __bf16* l) {
  unsigned loff = (unsigned)(unsigned long long)l;
  asm volatile("global_load_async_to_lds_b128 %0, %1, off"
               :: "v"(loff), "v"(g)
               : "memory");
}

__device__ __forceinline__ void wait_async0() {
#if __has_builtin(__builtin_amdgcn_s_wait_asynccnt)
  __builtin_amdgcn_s_wait_asynccnt(0);
#else
  asm volatile("s_wait_asynccnt 0x0" ::: "memory");
#endif
}

// ---------------- kernel 1: f32 -> bf16 conversions (+ weight transposes) ------

__global__ void convert_kernel(const float* __restrict__ x, const float* __restrict__ Wq,
                               const float* __restrict__ Wkv, const float* __restrict__ Wo,
                               __bf16* __restrict__ x_bf, __bf16* __restrict__ WqT,
                               __bf16* __restrict__ WkvT, __bf16* __restrict__ WoT) {
  int stride = gridDim.x * blockDim.x;
  int t0 = blockIdx.x * blockDim.x + threadIdx.x;
  for (int i = t0; i < NROWS * DIM; i += stride) x_bf[i] = (__bf16)x[i];
  for (int i = t0; i < DIM * INNER; i += stride) {
    int k = i / INNER, n = i % INNER;
    WqT[(size_t)n * DIM + k] = (__bf16)Wq[i];
  }
  for (int i = t0; i < DIM * 2 * DHEAD; i += stride) {
    int k = i / (2 * DHEAD), n = i % (2 * DHEAD);
    WkvT[(size_t)n * DIM + k] = (__bf16)Wkv[i];
  }
  for (int i = t0; i < INNER * DIM; i += stride) {
    int k = i / DIM, n = i % DIM;
    WoT[(size_t)n * INNER + k] = (__bf16)Wo[i];
  }
}

// ---------------- kernel 2: fused q/k/v projection  ---------------------------
// GEMM: [4096 x 1024] @ [1024 x 640]  -> q (scaled, bf16), k (bf16), v^T (bf16)

__global__ void proj_kernel(const __bf16* __restrict__ x_bf, const __bf16* __restrict__ WqT,
                            const __bf16* __restrict__ WkvT, __bf16* __restrict__ q_bf,
                            __bf16* __restrict__ k_bf, __bf16* __restrict__ vT_bf) {
  int lane = threadIdx.x & 31;
  int wave = threadIdx.x >> 5;
  int tile = blockIdx.x * 4 + wave;       // 256 * 40 tiles
  int tm = tile / 40, tn = tile % 40;
  int row0 = tm * 16, col0 = tn * 16;
  int lc = lane & 15, half = lane >> 4;

  const __bf16* abase = x_bf + (size_t)(row0 + lc) * DIM;
  const __bf16* btcol;
  if (col0 < INNER) btcol = WqT + (size_t)(col0 + lc) * DIM;
  else              btcol = WkvT + (size_t)(col0 - INNER + lc) * DIM;

  v8f c = {};
  for (int k0 = 0; k0 < DIM; k0 += 32)
    c = wmma_bf16(load_frag_a(abase, k0, lane), load_frag_bt(btcol, k0, lane), c);

#pragma unroll
  for (int r = 0; r < 8; ++r) {
    int row = row0 + r + 8 * half;
    int col = col0 + lc;
    float v = c[r];
    if (col < INNER) {
      q_bf[(size_t)row * INNER + col] = (__bf16)(v * QK_SCALE);
    } else if (col < INNER + DHEAD) {
      k_bf[(size_t)row * DHEAD + (col - INNER)] = (__bf16)v;
    } else {
      int d = col - INNER - DHEAD;
      int bb = row / SEQ, i = row % SEQ;
      vT_bf[((size_t)bb * DHEAD + d) * SEQ + i] = (__bf16)v;
    }
  }
}

// ---------------- kernel 3: flash attention (causal, +bias) -------------------
// grid (n/64, heads, batch), 128 threads; each wave owns a 16-row query tile.
// K/V tiles cooperatively staged into LDS with async copies, double-buffered.

__global__ void attn_kernel(const __bf16* __restrict__ q_bf, const __bf16* __restrict__ k_bf,
                            const __bf16* __restrict__ vT_bf, const float* __restrict__ bias,
                            __bf16* __restrict__ attn_bf) {
  __shared__ __bf16 ktile[2][32 * 64];   // [j-local][d]
  __shared__ __bf16 vtile[2][64 * 32];   // [d][j-local]
  __shared__ __bf16 lds_p[4][16 * 32];   // per-wave P remap (C-layout -> A-layout)

  int tid = threadIdx.x;
  int lane = tid & 31;
  int wave = tid >> 5;
  int lc = lane & 15, half = lane >> 4;
  int h = blockIdx.y, bb = blockIdx.z;
  int iblock = blockIdx.x * 64;
  int ibase = iblock + wave * 16;

  const __bf16* qrow = q_bf + ((size_t)(bb * SEQ) + ibase + lc) * INNER + h * DHEAD;
  v16bf aq0 = load_frag_a(qrow, 0, lane);
  v16bf aq1 = load_frag_a(qrow, 32, lane);

  const __bf16* kb = k_bf + (size_t)bb * SEQ * DHEAD;
  const __bf16* vb = vT_bf + (size_t)bb * DHEAD * SEQ;
  const float* biasb = bias + (size_t)h * SEQ * SEQ;

  // issue async copies for K tile (32x64) and V^T tile (64x32) at column j0
  auto stage = [&](int buf, int j0) {
#pragma unroll
    for (int c0 = 0; c0 < 2; ++c0) {
      int c = tid + c0 * 128;                      // 256 16B-chunks per tile
      int jl = c >> 3, ok = (c & 7) << 3;          // K tile: 8 chunks per row
      async_copy16(kb + (size_t)(j0 + jl) * DHEAD + ok, &ktile[buf][jl * 64 + ok]);
      int d = c >> 2, ov = (c & 3) << 3;           // V tile: 4 chunks per row
      async_copy16(vb + (size_t)d * SEQ + j0 + ov, &vtile[buf][d * 32 + ov]);
    }
  };

  v8f o0 = {}, o1 = {}, o2 = {}, o3 = {};
  float m[8], l[8];
#pragma unroll
  for (int r = 0; r < 8; ++r) { m[r] = NEG_BIG; l[r] = 0.f; }

  int jmax = iblock + 64;                 // uniform across block (causal bound of top wave)
  stage(0, 0);
  wait_async0();
  __syncthreads();

  int buf = 0;
  for (int j0 = 0; j0 < jmax; j0 += 32) {
    if (j0 + 32 < jmax) stage(buf ^ 1, j0 + 32);   // prefetch next tile (overlapped)

    // ---- S = Q K^T (16x32 scores as two 16x16 accumulators, K=64 chained)
    const __bf16* kc0 = &ktile[buf][lc * 64];
    const __bf16* kc1 = &ktile[buf][(lc + 16) * 64];
    v8f z = {};
    v8f s0 = wmma_bf16(aq0, load_frag_bt(kc0, 0, lane), z);
    s0 = wmma_bf16(aq1, load_frag_bt(kc0, 32, lane), s0);
    v8f s1 = wmma_bf16(aq0, load_frag_bt(kc1, 0, lane), z);
    s1 = wmma_bf16(aq1, load_frag_bt(kc1, 32, lane), s1);

    // ---- bias + causal mask + online softmax
    // (tiles fully beyond a wave's causal bound are no-ops: p=0, rescale=1)
#pragma unroll
    for (int r = 0; r < 8; ++r) {
      int i = ibase + r + 8 * half;
      int j = j0 + lc;
      float v0 = s0[r] + biasb[(size_t)i * SEQ + j];
      float v1 = s1[r] + biasb[(size_t)i * SEQ + j + 16];
      if (j > i) v0 = NEG_BIG;
      if (j + 16 > i) v1 = NEG_BIG;
      float mn = fmaxf(m[r], rowmax16(fmaxf(v0, v1)));
      float p0 = __expf(v0 - mn);
      float p1 = __expf(v1 - mn);
      float fr = __expf(m[r] - mn);
      l[r] = l[r] * fr + rowsum16(p0 + p1);
      m[r] = mn;
      o0[r] *= fr; o1[r] *= fr; o2[r] *= fr; o3[r] *= fr;
      __bf16* lp = &lds_p[wave][(r + 8 * half) * 32];
      lp[lc] = (__bf16)p0;
      lp[lc + 16] = (__bf16)p1;
    }
    asm volatile("s_wait_dscnt 0x0" ::: "memory");   // intra-wave DS ordering

    // ---- P (16x32) in A-fragment layout from LDS
    v16bf ap = load_frag_a(&lds_p[wave][lc * 32], 0, lane);

    // ---- O += P V  (V^T tile in LDS, K(=j)-contiguous per d column)
    o0 = wmma_bf16(ap, load_frag_bt(&vtile[buf][(0 * 16 + lc) * 32], 0, lane), o0);
    o1 = wmma_bf16(ap, load_frag_bt(&vtile[buf][(1 * 16 + lc) * 32], 0, lane), o1);
    o2 = wmma_bf16(ap, load_frag_bt(&vtile[buf][(2 * 16 + lc) * 32], 0, lane), o2);
    o3 = wmma_bf16(ap, load_frag_bt(&vtile[buf][(3 * 16 + lc) * 32], 0, lane), o3);

    wait_async0();        // own prefetches landed
    __syncthreads();      // everyone done reading tile[buf] / writing tile[buf^1]
    buf ^= 1;
  }

  // ---- normalize and store (b, n, INNER) bf16
  __bf16* outp = attn_bf + ((size_t)(bb * SEQ) + ibase) * INNER + h * DHEAD;
#pragma unroll
  for (int r = 0; r < 8; ++r) {
    float inv = 1.f / l[r];
    int row = r + 8 * half;
    outp[(size_t)row * INNER + 0 * 16 + lc] = (__bf16)(o0[r] * inv);
    outp[(size_t)row * INNER + 1 * 16 + lc] = (__bf16)(o1[r] * inv);
    outp[(size_t)row * INNER + 2 * 16 + lc] = (__bf16)(o2[r] * inv);
    outp[(size_t)row * INNER + 3 * 16 + lc] = (__bf16)(o3[r] * inv);
  }
}

// ---------------- kernel 4: output projection + bias --------------------------
// [4096 x 512] @ [512 x 1024] + bo -> f32

__global__ void outproj_kernel(const __bf16* __restrict__ attn_bf, const __bf16* __restrict__ WoT,
                               const float* __restrict__ bo, float* __restrict__ out) {
  int lane = threadIdx.x & 31;
  int wave = threadIdx.x >> 5;
  int tile = blockIdx.x * 4 + wave;       // 256 * 64 tiles
  int tm = tile / 64, tn = tile % 64;
  int row0 = tm * 16, col0 = tn * 16;
  int lc = lane & 15, half = lane >> 4;

  const __bf16* abase = attn_bf + (size_t)(row0 + lc) * INNER;
  const __bf16* btcol = WoT + (size_t)(col0 + lc) * INNER;

  v8f c = {};
  for (int k0 = 0; k0 < INNER; k0 += 32)
    c = wmma_bf16(load_frag_a(abase, k0, lane), load_frag_bt(btcol, k0, lane), c);

#pragma unroll
  for (int r = 0; r < 8; ++r) {
    int row = row0 + r + 8 * half;
    int col = col0 + lc;
    out[(size_t)row * DIM + col] = c[r] + bo[col];
  }
}

// ---------------- launcher ----------------------------------------------------

extern "C" void kernel_launch(void* const* d_in, const int* in_sizes, int n_in,
                              void* d_out, int out_size, void* d_ws, size_t ws_size,
                              hipStream_t stream) {
  const float* x    = (const float*)d_in[0];
  const float* bias = (const float*)d_in[1];
  const float* Wq   = (const float*)d_in[2];
  const float* Wkv  = (const float*)d_in[3];
  const float* Wo   = (const float*)d_in[4];
  const float* bo   = (const float*)d_in[5];
  float* out = (float*)d_out;

  char* ws = (char*)d_ws;
  __bf16* x_bf  = (__bf16*)(ws + 0);                 // 8 MB
  __bf16* WqT   = (__bf16*)(ws + 8388608);           // 1 MB
  __bf16* WkvT  = (__bf16*)(ws + 9437184);           // 256 KB
  __bf16* WoT   = (__bf16*)(ws + 9699328);           // 1 MB
  __bf16* q_bf  = (__bf16*)(ws + 10747904);          // 4 MB
  __bf16* k_bf  = (__bf16*)(ws + 14942208);          // 512 KB
  __bf16* vT_bf = (__bf16*)(ws + 15466496);          // 512 KB
  __bf16* a_bf  = (__bf16*)(ws + 15990784);          // 4 MB   (total ~19.3 MB)

  convert_kernel<<<2048, 256, 0, stream>>>(x, Wq, Wkv, Wo, x_bf, WqT, WkvT, WoT);
  proj_kernel<<<2560, 128, 0, stream>>>(x_bf, WqT, WkvT, q_bf, k_bf, vT_bf);
  attn_kernel<<<dim3(SEQ / 64, NHEADS, BATCH), 128, 0, stream>>>(q_bf, k_bf, vT_bf, bias, a_bf);
  outproj_kernel<<<4096, 128, 0, stream>>>(a_bf, WoT, bo, out);
}
